// SelfAttention_27238682591985
// MI455X (gfx1250) — matrix operand order repaired and checked by reading, not compiled, not run on hardware
//
#include <hip/hip_runtime.h>

// ---------------- problem constants ----------------
#define BB 4
#define TT 2048
#define CC 1024
#define HH 16
#define DH 64            // head dim
#define MROWS (BB * TT)  // 8192
#define N_QKV (3 * CC)   // 3072

typedef __attribute__((ext_vector_type(16))) __bf16 v16bf;
typedef __attribute__((ext_vector_type(8)))  float  v8f;

union ABPack { v16bf v; unsigned short u[16]; };
union CPack  { v8f   v; float          f[8];  };

// ---- CDNA5 async global->LDS copy (ASYNCcnt) ------------------------------
// global_load_async_to_lds_b128: VDST = LDS byte address (VGPR),
// VADDR = 64-bit global address (VGPR pair). ISA 15.18.3 opcode 98.
#define ASYNC_LD_B128(ldsaddr, gaddr)                                          \
    asm volatile("global_load_async_to_lds_b128 %0, %1, off"                   \
                 :: "v"(ldsaddr), "v"(gaddr) : "memory")

#if defined(__has_builtin)
#if __has_builtin(__builtin_amdgcn_s_wait_asynccnt)
#define WAIT_ASYNC(n) __builtin_amdgcn_s_wait_asynccnt(n)
#endif
#endif
#ifndef WAIT_ASYNC
#define WAIT_ASYNC(n) asm volatile("s_wait_asynccnt %0" :: "i"(n) : "memory")
#endif

__device__ __forceinline__ unsigned short f32_to_bf16(float f) {
    union { float f; unsigned int u; } c; c.f = f;
    unsigned int u = c.u;
    u += 0x7fffu + ((u >> 16) & 1u);          // round to nearest even
    return (unsigned short)(u >> 16);
}

// ---------------- fp32 -> bf16 convert ----------------
__global__ void k_cvt_bf16(const float* __restrict__ src,
                           unsigned short* __restrict__ dst, int n) {
    int i = blockIdx.x * blockDim.x + threadIdx.x;
    int stride = gridDim.x * blockDim.x;
    for (; i < n; i += stride) dst[i] = f32_to_bf16(src[i]);
}

// ---------------- bf16 GEMM: Y[m,n] = sum_k X[m,k] * W[n,k] (+bias) --------
// block = 128 threads (4 waves). Block tile: 64(M) x 64(N). K step = 32.
// Double-buffered LDS filled by async global->LDS copies; WMMA on the other
// buffer overlaps the in-flight copy of the next K tile.
__global__ void k_gemm_bf16(const unsigned short* __restrict__ X,
                            const unsigned short* __restrict__ W,
                            unsigned short* __restrict__ Yb,
                            float* __restrict__ Yf,
                            const float* __restrict__ bias,
                            int M, int N, int K) {
    __shared__ unsigned short Xs[2][64][40];   // 80B row stride (16B aligned)
    __shared__ unsigned short Ws[2][64][40];

    const int tid  = threadIdx.x;
    const int wv   = tid >> 5;
    const int lane = tid & 31;
    const int l16  = lane & 15;
    const int lhi  = lane >> 4;                // lane half (0/1)
    const int m0   = blockIdx.y * 64;
    const int n0   = blockIdx.x * 64;
    // staging assignment: rows r0 and r0+32, 16B segment sg (in ushorts)
    const int r0 = tid >> 2;
    const int sg = (tid & 3) * 8;

    CPack acc[4];
#pragma unroll
    for (int t = 0; t < 4; ++t)
#pragma unroll
        for (int r = 0; r < 8; ++r) acc[t].f[r] = 0.0f;

    auto issue_tile = [&](int buf, int k0) {
        unsigned long long gx0 = (unsigned long long)(size_t)&X[(size_t)(m0 + r0)      * K + k0 + sg];
        unsigned long long gx1 = (unsigned long long)(size_t)&X[(size_t)(m0 + r0 + 32) * K + k0 + sg];
        unsigned long long gw0 = (unsigned long long)(size_t)&W[(size_t)(n0 + r0)      * K + k0 + sg];
        unsigned long long gw1 = (unsigned long long)(size_t)&W[(size_t)(n0 + r0 + 32) * K + k0 + sg];
        ASYNC_LD_B128((unsigned)(size_t)&Xs[buf][r0][sg],      gx0);
        ASYNC_LD_B128((unsigned)(size_t)&Xs[buf][r0 + 32][sg], gx1);
        ASYNC_LD_B128((unsigned)(size_t)&Ws[buf][r0][sg],      gw0);
        ASYNC_LD_B128((unsigned)(size_t)&Ws[buf][r0 + 32][sg], gw1);
    };

    const int ntiles = K >> 5;
    issue_tile(0, 0);

    for (int t = 0; t < ntiles; ++t) {
        if (t + 1 < ntiles) {                   // overlap next copy with WMMA
            issue_tile((t + 1) & 1, (t + 1) * 32);
            WAIT_ASYNC(4);                      // tile t's 4 copies complete
        } else {
            WAIT_ASYNC(0);
        }
        __syncthreads();

        const unsigned short (*xs)[40] = Xs[t & 1];
        const unsigned short (*ws)[40] = Ws[t & 1];

        // A pack: 16x32 bf16 (lane = M, slots hold K per ISA layout)
        ABPack a;
#pragma unroll
        for (int h = 0; h < 16; ++h) {
            int k = 16 * (h >> 3) + 8 * lhi + (h & 7);
            a.u[h] = xs[16 * wv + l16][k];
        }
        // B packs: 32x16 bf16 (lane = N, slots hold K)
#pragma unroll
        for (int tt = 0; tt < 4; ++tt) {
            ABPack b;
#pragma unroll
            for (int h = 0; h < 16; ++h) {
                b.u[h] = ws[tt * 16 + l16][16 * (h >> 3) + 8 * lhi + (h & 7)];
            }
            acc[tt].v = __builtin_amdgcn_wmma_f32_16x16x32_bf16(
                false, a.v, false, b.v, (short)0, acc[tt].v, false, false);
        }
        __syncthreads();                        // reads done before overwrite
    }

    // epilogue: C/D layout (m = r + 8*lane_half, n = lane&15)
#pragma unroll
    for (int t = 0; t < 4; ++t) {
#pragma unroll
        for (int r = 0; r < 8; ++r) {
            int row = m0 + 16 * wv + r + 8 * lhi;
            int col = n0 + t * 16 + l16;
            float v = acc[t].f[r];
            if (Yf) {
                Yf[(size_t)row * N + col] = v + (bias ? bias[col] : 0.0f);
            } else {
                Yb[(size_t)row * N + col] = f32_to_bf16(v);
            }
        }
    }
}

// ---------------- flash attention (causal) ----------------
// grid = (T/64, H, B), block = 128 (4 waves). Wave w: 16 queries, full 64-d O.
// 32-key K/V tiles double-buffered in LDS via async copies.
__global__ void k_flash_attn(const unsigned short* __restrict__ qkv,
                             unsigned short* __restrict__ attn) {
    __shared__ unsigned short Ks[2][32][72];    // 144B stride, 16B aligned
    __shared__ unsigned short Vs[2][32][72];
    __shared__ unsigned short Ps[4][16][40];    // per-wave P transpose buffer

    const int tid  = threadIdx.x;
    const int wv   = tid >> 5;
    const int lane = tid & 31;
    const int l16  = lane & 15;
    const int lhi  = lane >> 4;
    const int chunk = blockIdx.x;
    const int hh    = blockIdx.y;
    const int b     = blockIdx.z;
    const int qbase = chunk * 64;
    const size_t rowstride = (size_t)N_QKV;
    const size_t baseq = (size_t)b * TT * rowstride;
    // staging: rows r0 and r0+16, 16B segment sg
    const int r0 = tid >> 3;
    const int sg = (tid & 7) * 8;

    auto issue_kv = [&](int buf, int j) {
        size_t g0 = baseq + (size_t)(j + r0)      * rowstride + hh * DH + sg;
        size_t g1 = baseq + (size_t)(j + r0 + 16) * rowstride + hh * DH + sg;
        ASYNC_LD_B128((unsigned)(size_t)&Ks[buf][r0][sg],
                      (unsigned long long)(size_t)&qkv[g0 + CC]);
        ASYNC_LD_B128((unsigned)(size_t)&Ks[buf][r0 + 16][sg],
                      (unsigned long long)(size_t)&qkv[g1 + CC]);
        ASYNC_LD_B128((unsigned)(size_t)&Vs[buf][r0][sg],
                      (unsigned long long)(size_t)&qkv[g0 + 2 * CC]);
        ASYNC_LD_B128((unsigned)(size_t)&Vs[buf][r0 + 16][sg],
                      (unsigned long long)(size_t)&qkv[g1 + 2 * CC]);
    };

    // Q tile (16x64) for this wave -> two A packs (d = 0..31, 32..63)
    ABPack aq[2];
    {
        const int q = qbase + 16 * wv + l16;
        const size_t qoff = baseq + (size_t)q * rowstride + hh * DH;
#pragma unroll
        for (int c = 0; c < 2; ++c)
#pragma unroll
            for (int h = 0; h < 16; ++h) {
                int d = c * 32 + 16 * (h >> 3) + 8 * lhi + (h & 7);
                aq[c].u[h] = qkv[qoff + d];
            }
    }

    CPack oacc[4];
    float mrow[8], lrow[8];
#pragma unroll
    for (int t = 0; t < 4; ++t)
#pragma unroll
        for (int r = 0; r < 8; ++r) oacc[t].f[r] = 0.0f;
#pragma unroll
    for (int r = 0; r < 8; ++r) { mrow[r] = -1e30f; lrow[r] = 0.0f; }

    const float scale = 0.125f;                 // 1/sqrt(64)
    const int nblocks = chunk * 2 + 2;          // keys processed: qbase+64

    issue_kv(0, 0);

    for (int jb = 0; jb < nblocks; ++jb) {
        const int j = jb * 32;
        if (jb + 1 < nblocks) {
            issue_kv((jb + 1) & 1, (jb + 1) * 32);
            WAIT_ASYNC(4);
        } else {
            WAIT_ASYNC(0);
        }
        __syncthreads();

        const unsigned short (*ks)[72] = Ks[jb & 1];
        const unsigned short (*vs)[72] = Vs[jb & 1];

        // S = Q K^T : two 16-key groups, K-dim 64 via two chained WMMAs
        CPack s[2];
#pragma unroll
        for (int g = 0; g < 2; ++g) {
#pragma unroll
            for (int r = 0; r < 8; ++r) s[g].f[r] = 0.0f;
#pragma unroll
            for (int c = 0; c < 2; ++c) {
                ABPack bk;
#pragma unroll
                for (int h = 0; h < 16; ++h) {
                    bk.u[h] = ks[g * 16 + l16][c * 32 + 16 * (h >> 3) + 8 * lhi + (h & 7)];
                }
                s[g].v = __builtin_amdgcn_wmma_f32_16x16x32_bf16(
                    false, aq[c].v, false, bk.v, (short)0, s[g].v, false, false);
            }
        }

        // online softmax per query row; write P (bf16) transposed via LDS
#pragma unroll
        for (int r = 0; r < 8; ++r) {
            const int mloc = r + 8 * lhi;
            const int q = qbase + 16 * wv + mloc;
            const int key0 = j + l16;
            const int key1 = j + 16 + l16;
            float s0 = s[0].f[r] * scale;
            float s1 = s[1].f[r] * scale;
            if (key0 > q) s0 = -1e30f;
            if (key1 > q) s1 = -1e30f;
            float mx = fmaxf(s0, s1);
#pragma unroll
            for (int off = 1; off < 16; off <<= 1)
                mx = fmaxf(mx, __shfl_xor(mx, off, 32));
            const float mnew  = fmaxf(mrow[r], mx);
            const float alpha = __expf(mrow[r] - mnew);
            const float p0 = __expf(s0 - mnew);
            const float p1 = __expf(s1 - mnew);
            float rs = p0 + p1;
#pragma unroll
            for (int off = 1; off < 16; off <<= 1)
                rs += __shfl_xor(rs, off, 32);
            lrow[r] = lrow[r] * alpha + rs;
            mrow[r] = mnew;
#pragma unroll
            for (int t = 0; t < 4; ++t) oacc[t].f[r] *= alpha;
            Ps[wv][mloc][l16]      = f32_to_bf16(p0);
            Ps[wv][mloc][16 + l16] = f32_to_bf16(p1);
        }
        // wave-local LDS RAW fence (per-wave LDS ops are in order)
        asm volatile("s_wait_dscnt 0" ::: "memory");

        // O += P V : A = P (16x32), B = V (32x16) per 16-wide d group
        ABPack ap;
#pragma unroll
        for (int h = 0; h < 16; ++h) {
            int k = 16 * (h >> 3) + 8 * lhi + (h & 7);
            ap.u[h] = Ps[wv][l16][k];
        }
#pragma unroll
        for (int t = 0; t < 4; ++t) {
            ABPack bv;
#pragma unroll
            for (int h = 0; h < 16; ++h) {
                bv.u[h] = vs[16 * (h >> 3) + 8 * lhi + (h & 7)][t * 16 + l16];
            }
            oacc[t].v = __builtin_amdgcn_wmma_f32_16x16x32_bf16(
                false, ap.v, false, bv.v, (short)0, oacc[t].v, false, false);
        }
        __syncthreads();                        // reads done before overwrite
    }

    // normalize and store O tile (bf16) into attn buffer [B*T, C]
#pragma unroll
    for (int r = 0; r < 8; ++r) {
        const int mloc = r + 8 * lhi;
        const int q = qbase + 16 * wv + mloc;
        const float inv = (lrow[r] > 0.0f) ? (1.0f / lrow[r]) : 0.0f;
#pragma unroll
        for (int t = 0; t < 4; ++t) {
            attn[(size_t)(b * TT + q) * CC + hh * DH + t * 16 + l16] =
                f32_to_bf16(oacc[t].f[r] * inv);
        }
    }
}

// ---------------- launcher ----------------
extern "C" void kernel_launch(void* const* d_in, const int* in_sizes, int n_in,
                              void* d_out, int out_size, void* d_ws, size_t ws_size,
                              hipStream_t stream) {
    const float* x      = (const float*)d_in[0];   // [B,T,C]
    const float* w_attn = (const float*)d_in[1];   // [3C,C]
    const float* w_proj = (const float*)d_in[2];   // [C,C]
    const float* b_proj = (const float*)d_in[3];   // [C]
    float* out = (float*)d_out;                    // [B,T,C] fp32

    // workspace layout (bf16 elements): xb | wab | wpb | qkv | attn  (~88 MB)
    unsigned short* base = (unsigned short*)d_ws;
    unsigned short* xb   = base;
    unsigned short* wab  = xb  + (size_t)MROWS * CC;
    unsigned short* wpb  = wab + (size_t)N_QKV * CC;
    unsigned short* qkv  = wpb + (size_t)CC * CC;
    unsigned short* attn = qkv + (size_t)MROWS * N_QKV;

    // 1) fp32 -> bf16
    k_cvt_bf16<<<4096, 256, 0, stream>>>(x,      xb,  MROWS * CC);
    k_cvt_bf16<<<4096, 256, 0, stream>>>(w_attn, wab, N_QKV * CC);
    k_cvt_bf16<<<2048, 256, 0, stream>>>(w_proj, wpb, CC * CC);

    // 2) qkv = x @ w_attn^T   (M=8192, N=3072, K=1024) -> bf16
    k_gemm_bf16<<<dim3(N_QKV / 64, MROWS / 64), 128, 0, stream>>>(
        xb, wab, qkv, nullptr, nullptr, MROWS, N_QKV, CC);

    // 3) flash attention -> attn (bf16, [B*T, C])
    k_flash_attn<<<dim3(TT / 64, HH, BB), 128, 0, stream>>>(qkv, attn);

    // 4) out = attn @ w_proj^T + b_proj  (fp32 out)
    k_gemm_bf16<<<dim3(CC / 64, MROWS / 64), 128, 0, stream>>>(
        attn, wpb, nullptr, out, b_proj, MROWS, CC, CC);
}
